// Attention_63995012710903
// MI455X (gfx1250) — matrix operand order repaired and checked by reading, not compiled, not run on hardware
//
#include <hip/hip_runtime.h>
#include <hip/hip_bf16.h>

// ---------------- problem constants (match reference) ----------------
#define BATCH   2
#define SEQ     2048
#define DMODEL  512
#define NHEADS  8
#define DHEAD   64
#define CHUNKS  8
#define CROWS   (SEQ / CHUNKS)   // 256 rows per suffix-scan chunk

#define NROWT   (BATCH * SEQ / 16)       // 256 row tiles
#define KSTEPS  (DMODEL / 32)            // 16 K-steps per GEMM
// fragment-major strides (in bf16 elements)
#define FRAG_E      16                   // elements per lane per fragment
#define FRAG_SZ     (32 * FRAG_E)        // 512 elems per 32-lane fragment (1 KB)
#define BGRP_SZ     (KSTEPS * 4 * FRAG_SZ)   // 32768 elems per 64-row weight group
#define AROW_SZ     (KSTEPS * FRAG_SZ)       // 8192 elems per 16-row A tile

// ---------------- CDNA5 WMMA types ----------------
typedef __attribute__((ext_vector_type(16))) __bf16 v16bf;
typedef __attribute__((ext_vector_type(8)))  __bf16 v8bf;
typedef __attribute__((ext_vector_type(8)))  float  v8f;

// fp32 -> bf16 round-to-nearest-even
static __device__ __forceinline__ __bf16 f2bf(float f) {
    unsigned u = __builtin_bit_cast(unsigned, f);
    unsigned r = u + 0x7fffu + ((u >> 16) & 1u);
    unsigned short h = (unsigned short)(r >> 16);
    return __builtin_bit_cast(__bf16, h);
}

static __device__ __forceinline__ v16bf cvt16(float4 a, float4 b, float4 c, float4 d) {
    v16bf v;
    v[0]  = f2bf(a.x); v[1]  = f2bf(a.y); v[2]  = f2bf(a.z); v[3]  = f2bf(a.w);
    v[4]  = f2bf(b.x); v[5]  = f2bf(b.y); v[6]  = f2bf(b.z); v[7]  = f2bf(b.w);
    v[8]  = f2bf(c.x); v[9]  = f2bf(c.y); v[10] = f2bf(c.z); v[11] = f2bf(c.w);
    v[12] = f2bf(d.x); v[13] = f2bf(d.y); v[14] = f2bf(d.z); v[15] = f2bf(d.w);
    return v;
}

// A-fragment from a row-major bf16 row (used for Z in the output GEMM):
// lane holds K = {d0..d0+7} and {d0+16..d0+23}, d0 = kb*32 + 8*half.
static __device__ __forceinline__ v16bf load_afrag_rowmajor(const __bf16* __restrict__ rowp, int d0) {
    v8bf lo = *reinterpret_cast<const v8bf*>(rowp + d0);
    v8bf hi = *reinterpret_cast<const v8bf*>(rowp + d0 + 16);
    return __builtin_shufflevector(lo, hi, 0, 1, 2, 3, 4, 5, 6, 7,
                                           8, 9, 10, 11, 12, 13, 14, 15);
}

static __device__ __forceinline__ v16bf ldfrag(const __bf16* __restrict__ p) {
    return *reinterpret_cast<const v16bf*>(p);                  // 32B, lane-contiguous
}

static __device__ __forceinline__ v8f wmma_bf16(v16bf a, v16bf b, v8f c) {
    return __builtin_amdgcn_wmma_f32_16x16x32_bf16(
        /*neg_a=*/false, a, /*neg_b=*/false, b,
        /*c_mod=*/(short)0, c, /*reuse_a=*/false, /*reuse_b=*/false);
}

// =====================================================================
// Kernel 0a: x (fp32 row-major) -> A-fragment-major bf16:
//   Af[rt][kb][lane][e], one thread per (rt, kb, lane).
// Writes are lane-contiguous 32B (perfectly coalesced).
// =====================================================================
__global__ void __launch_bounds__(256)
swizzleA_kernel(const float* __restrict__ src, __bf16* __restrict__ dst) {
    const int gid  = blockIdx.x * 256 + threadIdx.x;    // NROWT*KSTEPS*32 total
    const int lane = gid & 31;
    const int kb   = (gid >> 5) & (KSTEPS - 1);
    const int rt   = gid >> 9;
    const int m    = lane & 15;
    const int half = lane >> 4;
    const float* row = src + ((size_t)rt * 16 + m) * DMODEL + kb * 32 + 8 * half;
    const float4* p0 = reinterpret_cast<const float4*>(row);
    const float4* p1 = reinterpret_cast<const float4*>(row + 16);
    *reinterpret_cast<v16bf*>(dst + (size_t)gid * FRAG_E) = cvt16(p0[0], p0[1], p1[0], p1[1]);
}

// =====================================================================
// Kernel 0b: weight (fp32 [rows, DMODEL], rows multiple of 64) ->
//   B-fragment-major bf16: Bf[g][kb][t][lane][e], g = 64-row group.
// One thread per (g, kb, t, lane); lane holds K = {k0..k0+7} and {k0+16..k0+23}
// with k0 = kb*32 + 8*half.
// =====================================================================
__global__ void __launch_bounds__(256)
swizzleB_kernel(const float* __restrict__ src, __bf16* __restrict__ dst) {
    const int gid  = blockIdx.x * 256 + threadIdx.x;    // (rows/64)*KSTEPS*4*32 total
    const int lane = gid & 31;
    const int t    = (gid >> 5) & 3;
    const int kb   = (gid >> 7) & (KSTEPS - 1);
    const int g    = gid >> 11;
    const int m    = lane & 15;
    const int half = lane >> 4;
    const int n    = g * 64 + t * 16 + m;
    const float* row = src + (size_t)n * DMODEL + kb * 32 + 8 * half;
    const float4* p0 = reinterpret_cast<const float4*>(row);
    const float4* p1 = reinterpret_cast<const float4*>(row + 16);
    *reinterpret_cast<v16bf*>(dst + (size_t)gid * FRAG_E) = cvt16(p0[0], p0[1], p1[0], p1[1]);
}

// =====================================================================
// Kernel 1: fused Q/K/V projection (fragment-major bf16 operands).
// grid = NROWT, block = 32x8; wave y handles head y for one 16-row tile.
// Every load is base + lane*32B + literal offset.
// =====================================================================
__global__ void __launch_bounds__(256)
qkv_proj_kernel(const __bf16* __restrict__ Af,
                const __bf16* __restrict__ Wq,
                const __bf16* __restrict__ Wk,
                const __bf16* __restrict__ Wv,
                float* __restrict__ Q, float* __restrict__ K, float* __restrict__ V) {
    const int lane = threadIdx.x;          // 0..31 (wave32)
    const int h    = threadIdx.y;          // head

    const __bf16* ap = Af + (size_t)blockIdx.x * AROW_SZ + (size_t)lane * FRAG_E;
    const __bf16* qp = Wq + (size_t)h * BGRP_SZ + (size_t)lane * FRAG_E;
    const __bf16* kp = Wk + (size_t)h * BGRP_SZ + (size_t)lane * FRAG_E;
    const __bf16* vp = Wv + (size_t)h * BGRP_SZ + (size_t)lane * FRAG_E;

    const v8f z8 = {0.f, 0.f, 0.f, 0.f, 0.f, 0.f, 0.f, 0.f};
    v8f accQ[4], accK[4], accV[4];
    #pragma unroll
    for (int t = 0; t < 4; ++t) { accQ[t] = z8; accK[t] = z8; accV[t] = z8; }

    #pragma unroll 4
    for (int kb = 0; kb < KSTEPS; ++kb) {
        const v16bf af = ldfrag(ap + kb * FRAG_SZ);
        #pragma unroll
        for (int t = 0; t < 4; ++t) {
            const int o = (kb * 4 + t) * FRAG_SZ;      // literal instruction offsets
            accQ[t] = wmma_bf16(af, ldfrag(qp + o), accQ[t]);
            accK[t] = wmma_bf16(af, ldfrag(kp + o), accK[t]);
            accV[t] = wmma_bf16(af, ldfrag(vp + o), accV[t]);
        }
    }

    const int g0   = blockIdx.x * 16;
    const int b    = g0 / SEQ;
    const int srow = g0 % SEQ;
    const int m15  = lane & 15;
    const int half = lane >> 4;
    const size_t base = (((size_t)b * NHEADS + h) * SEQ + srow) * DHEAD;
    float* qo = Q + base;
    float* ko = K + base;
    float* vo = V + base;
    #pragma unroll
    for (int t = 0; t < 4; ++t) {
        const int n = t * 16 + m15;
        #pragma unroll
        for (int r = 0; r < 8; ++r) {
            const int mm = r + 8 * half;               // D-matrix row mapping
            qo[(size_t)mm * DHEAD + n] = accQ[t][r];
            ko[(size_t)mm * DHEAD + n] = accK[t][r];
            vo[(size_t)mm * DHEAD + n] = accV[t][r];
        }
    }
}

// =====================================================================
// Kernel 2: e[b,i,q] = exp( (q . k) / 8 ) — only the score diagonal
// survives the reference's mask in fp32.
// =====================================================================
__global__ void __launch_bounds__(256)
diag_exp_kernel(const float* __restrict__ Q, const float* __restrict__ K,
                float* __restrict__ E) {
    const int idx = blockIdx.x * 256 + threadIdx.x;    // [0, B*H*SEQ)
    const float4* qp = reinterpret_cast<const float4*>(Q + (size_t)idx * DHEAD);
    const float4* kp = reinterpret_cast<const float4*>(K + (size_t)idx * DHEAD);
    float s = 0.f;
    #pragma unroll
    for (int i = 0; i < DHEAD / 4; ++i) {
        float4 a = qp[i], b = kp[i];
        s += a.x * b.x + a.y * b.y + a.z * b.z + a.w * b.w;
    }
    E[idx] = expf(s * 0.125f);   // 1/sqrt(64)
}

// =====================================================================
// Kernel 3a: per-chunk column sums of V (phase 1 of parallel suffix scan)
// =====================================================================
__global__ void __launch_bounds__(64)
vchunk_sum_kernel(const float* __restrict__ V, float* __restrict__ CS) {
    const int t  = threadIdx.x;
    const int c  = blockIdx.x;
    const int bi = blockIdx.y;
    const float* vp = V + ((size_t)bi * SEQ + (size_t)c * CROWS) * DHEAD + t;
    float s = 0.f;
    for (int p = 0; p < CROWS; ++p) s += vp[(size_t)p * DHEAD];
    CS[((size_t)bi * CHUNKS + c) * DHEAD + t] = s;
}

// =====================================================================
// Kernel 3b: z[b,q,i*64+t] = (e_q * v[q,t] + suffix(v)) / (e_q + (S-1-q))
// =====================================================================
__global__ void __launch_bounds__(64)
zcombine_kernel(const float* __restrict__ V, const float* __restrict__ E,
                const float* __restrict__ CS, __bf16* __restrict__ Z) {
    const int t  = threadIdx.x;
    const int c  = blockIdx.x;
    const int bi = blockIdx.y;
    const int b  = bi / NHEADS;
    const int i  = bi % NHEADS;

    float run = 0.f;
    for (int cc = c + 1; cc < CHUNKS; ++cc)
        run += CS[((size_t)bi * CHUNKS + cc) * DHEAD + t];

    for (int p = (c + 1) * CROWS - 1; p >= c * CROWS; --p) {
        const float v = V[((size_t)bi * SEQ + p) * DHEAD + t];
        const float e = E[(size_t)bi * SEQ + p];
        const float z = (e * v + run) / (e + (float)(SEQ - 1 - p));
        Z[((size_t)b * SEQ + p) * DMODEL + (size_t)i * DHEAD + t] = f2bf(z);
        run += v;
    }
}

// =====================================================================
// Kernel 4: out[b,q,d] = sum_f W_O[d,f] * z[b,q,f]  (fragment-major W_O)
// grid = NROWT, block = 32x8; wave y covers output columns [64y, 64y+64).
// =====================================================================
__global__ void __launch_bounds__(256)
out_proj_kernel(const __bf16* __restrict__ Z, const __bf16* __restrict__ Wo,
                float* __restrict__ Out) {
    const int lane = threadIdx.x;
    const int w    = threadIdx.y;
    const int g0   = blockIdx.x * 16;
    const int m15  = lane & 15;
    const int half = lane >> 4;

    const __bf16* zrow = Z  + (size_t)(g0 + m15) * DMODEL;
    const __bf16* wp   = Wo + (size_t)w * BGRP_SZ + (size_t)lane * FRAG_E;

    const v8f z8 = {0.f, 0.f, 0.f, 0.f, 0.f, 0.f, 0.f, 0.f};
    v8f acc[4];
    #pragma unroll
    for (int t = 0; t < 4; ++t) acc[t] = z8;

    #pragma unroll 4
    for (int kb = 0; kb < KSTEPS; ++kb) {
        const v16bf af = load_afrag_rowmajor(zrow, kb * 32 + 8 * half);
        #pragma unroll
        for (int t = 0; t < 4; ++t)
            acc[t] = wmma_bf16(af, ldfrag(wp + (kb * 4 + t) * FRAG_SZ), acc[t]);
    }

    #pragma unroll
    for (int t = 0; t < 4; ++t) {
        const int n = w * 64 + t * 16 + m15;
        #pragma unroll
        for (int r = 0; r < 8; ++r)
            Out[(size_t)(g0 + r + 8 * half) * DMODEL + n] = acc[t][r];
    }
}

// =====================================================================
// Launch
// =====================================================================
extern "C" void kernel_launch(void* const* d_in, const int* in_sizes, int n_in,
                              void* d_out, int out_size, void* d_ws, size_t ws_size,
                              hipStream_t stream) {
    const float* x  = (const float*)d_in[0];
    const float* Wq = (const float*)d_in[1];
    const float* Wk = (const float*)d_in[2];
    const float* Wv = (const float*)d_in[3];
    const float* Wo = (const float*)d_in[4];
    float* out = (float*)d_out;

    // Workspace carve-up (~34.2 MB total)
    char* ws = (char*)d_ws;
    const size_t nqkv = (size_t)BATCH * NHEADS * SEQ * DHEAD;    // 2M elems
    const size_t nx   = (size_t)BATCH * SEQ * DMODEL;            // 2M elems
    const size_t nw   = (size_t)DHEAD * NHEADS * DMODEL;         // 256K elems
    float*  Q   = (float*)ws;   ws += nqkv * sizeof(float);
    float*  K   = (float*)ws;   ws += nqkv * sizeof(float);
    float*  V   = (float*)ws;   ws += nqkv * sizeof(float);
    float*  E   = (float*)ws;   ws += (size_t)BATCH * NHEADS * SEQ * sizeof(float);
    float*  CS  = (float*)ws;   ws += (size_t)BATCH * NHEADS * CHUNKS * DHEAD * sizeof(float);
    __bf16* Zb  = (__bf16*)ws;  ws += nx * sizeof(__bf16);
    __bf16* Af  = (__bf16*)ws;  ws += nx * sizeof(__bf16);       // A-fragment x
    __bf16* Wqb = (__bf16*)ws;  ws += nw * sizeof(__bf16);
    __bf16* Wkb = (__bf16*)ws;  ws += nw * sizeof(__bf16);
    __bf16* Wvb = (__bf16*)ws;  ws += nw * sizeof(__bf16);
    __bf16* Wob = (__bf16*)ws;

    // --- one-shot fp32 -> bf16 fragment-major swizzles ---
    swizzleA_kernel<<<dim3(NROWT * KSTEPS * 32 / 256), dim3(256), 0, stream>>>(x, Af);
    swizzleB_kernel<<<dim3((unsigned)(nw / FRAG_E / 256)), dim3(256), 0, stream>>>(Wq, Wqb);
    swizzleB_kernel<<<dim3((unsigned)(nw / FRAG_E / 256)), dim3(256), 0, stream>>>(Wk, Wkb);
    swizzleB_kernel<<<dim3((unsigned)(nw / FRAG_E / 256)), dim3(256), 0, stream>>>(Wv, Wvb);
    swizzleB_kernel<<<dim3((unsigned)(nw / FRAG_E / 256)), dim3(256), 0, stream>>>(Wo, Wob);

    const dim3 gemmBlk(32, 8);
    qkv_proj_kernel<<<dim3(NROWT), gemmBlk, 0, stream>>>(Af, Wqb, Wkb, Wvb, Q, K, V);
    diag_exp_kernel<<<dim3(BATCH * NHEADS * SEQ / 256), dim3(256), 0, stream>>>(Q, K, E);
    vchunk_sum_kernel<<<dim3(CHUNKS, BATCH * NHEADS), dim3(64), 0, stream>>>(V, CS);
    zcombine_kernel<<<dim3(CHUNKS, BATCH * NHEADS), dim3(64), 0, stream>>>(V, E, CS, Zb);
    out_proj_kernel<<<dim3(NROWT), gemmBlk, 0, stream>>>(Zb, Wob, out);
}
